// BANMO_81046032875689
// MI455X (gfx1250) — compile-verified, matrix operand order checked by reading, not run on hardware
//
#include <hip/hip_runtime.h>
#include <hip/hip_bf16.h>

typedef __attribute__((ext_vector_type(16))) _Float16 v16h;
typedef __attribute__((ext_vector_type(8)))  _Float16 v8h;
typedef __attribute__((ext_vector_type(8)))  float    v8f;

// ---------------- workspace layout ----------------
// f16 region (element offsets)
#define WS_W0   0            // 64 x 192
#define WS_W1   12288        // 64 x 64
#define WS_W2   16384
#define WS_W3   20480
#define WS_W4   24576        // 64 x 256
#define WS_WF   40960        // 64 x 64
#define WS_WD   45056        // 32 x 64
#define WS_WO   47104        // 32 x 32 (rows 25..31 zero)
#define WS_CODE 48128        // 128 (rest_pose_code f16)
#define WS_F16_TOTAL 48256
#define WS_F32_BYTES 96512   // f16 bytes rounded to 16
// f32 region (float offsets from WS_F32_BYTES)
#define FB0   0
#define FB1   64
#define FB2   128
#define FB3   192
#define FB4   256
#define FBF   320
#define FBD   384
#define FBO   416            // 32 (25..31 zero)
#define FBONE 448            // 25 * 27
#define FBONE_CNT 676        // 675 bone floats + alpha

// ======================================================================
// Kernel 1: weight conversion / padding to f16
// ======================================================================
struct PrepArgs {
  const float *w0, *w1, *w2, *w3, *w4, *wf, *wd, *wo;
  const float *b0, *b1, *b2, *b3, *b4, *bf, *bd, *bo;
  const float *rest;
};

__global__ void prep_kernel(PrepArgs pa, _Float16* ws16, float* wsf) {
  const int tid = blockIdx.x * blockDim.x + threadIdx.x;
  const int T   = gridDim.x * blockDim.x;
  for (int i = tid; i < 64*192; i += T) {           // W0: 64 x 191 -> 64 x 192
    int r = i / 192, c = i - r*192;
    ws16[WS_W0 + i] = (c < 191) ? (_Float16)pa.w0[r*191 + c] : (_Float16)0.f;
  }
  for (int i = tid; i < 64*64; i += T) ws16[WS_W1 + i] = (_Float16)pa.w1[i];
  for (int i = tid; i < 64*64; i += T) ws16[WS_W2 + i] = (_Float16)pa.w2[i];
  for (int i = tid; i < 64*64; i += T) ws16[WS_W3 + i] = (_Float16)pa.w3[i];
  for (int i = tid; i < 64*256; i += T) {           // W4: 64 x 255 -> 64 x 256
    int r = i >> 8, c = i & 255;
    ws16[WS_W4 + i] = (c < 255) ? (_Float16)pa.w4[r*255 + c] : (_Float16)0.f;
  }
  for (int i = tid; i < 64*64; i += T) ws16[WS_WF + i] = (_Float16)pa.wf[i];
  for (int i = tid; i < 32*64; i += T) ws16[WS_WD + i] = (_Float16)pa.wd[i];
  for (int i = tid; i < 32*32; i += T) {            // WO: 25 x 32 -> 32 x 32
    int r = i >> 5, c = i & 31;
    ws16[WS_WO + i] = (r < 25) ? (_Float16)pa.wo[r*32 + c] : (_Float16)0.f;
  }
  for (int i = tid; i < 128; i += T) ws16[WS_CODE + i] = (_Float16)pa.rest[i];
  for (int i = tid; i < 64; i += T) {
    wsf[FB0 + i] = pa.b0[i]; wsf[FB1 + i] = pa.b1[i]; wsf[FB2 + i] = pa.b2[i];
    wsf[FB3 + i] = pa.b3[i]; wsf[FB4 + i] = pa.b4[i]; wsf[FBF + i] = pa.bf[i];
  }
  for (int i = tid; i < 32; i += T) {
    wsf[FBD + i] = pa.bd[i];
    wsf[FBO + i] = (i < 25) ? pa.bo[i] : 0.f;
  }
}

// ======================================================================
// Kernel 2: pose path (tiny) — rt_head x2 + bone algebra
// ======================================================================
struct PoseArgs {
  const int*   tidx;
  const float* bones;
  const float* skin_aux;
  const float* rest_code;
  const float* basis_w;
  const float* basis_b;
  const float* w[8];
  const float* b[8];
  const float *wf, *bf, *wd, *bd, *wo, *bo;
};

__device__ __forceinline__ void so3_exp(const float* w, float R[9]) {
  float sq = fmaxf(w[0]*w[0] + w[1]*w[1] + w[2]*w[2], 1e-8f);
  float th = sqrtf(sq);
  float f1 = sinf(th) / th;
  float f2 = (1.f - cosf(th)) / sq;
  float K[9]  = {0.f, -w[2], w[1],  w[2], 0.f, -w[0],  -w[1], w[0], 0.f};
  float K2[9];
  for (int i = 0; i < 3; ++i)
    for (int j = 0; j < 3; ++j) {
      float a = 0.f;
      for (int k = 0; k < 3; ++k) a += K[i*3+k] * K[k*3+j];
      K2[i*3+j] = a;
    }
  for (int i = 0; i < 9; ++i)
    R[i] = ((i % 4) == 0 ? 1.f : 0.f) + f1*K[i] + f2*K2[i];
}

__device__ __forceinline__ void mat_to_quat(const float* m, float q[4]) {
  float m00=m[0],m01=m[1],m02=m[2],m10=m[3],m11=m[4],m12=m[5],m20=m[6],m21=m[7],m22=m[8];
  float qa[4];
  qa[0] = sqrtf(fmaxf(1.f + m00 + m11 + m22, 0.f));
  qa[1] = sqrtf(fmaxf(1.f + m00 - m11 - m22, 0.f));
  qa[2] = sqrtf(fmaxf(1.f - m00 + m11 - m22, 0.f));
  qa[3] = sqrtf(fmaxf(1.f - m00 - m11 + m22, 0.f));
  float cand[4][4] = {
    {qa[0]*qa[0], m21-m12, m02-m20, m10-m01},
    {m21-m12, qa[1]*qa[1], m10+m01, m02+m20},
    {m02-m20, m10+m01, qa[2]*qa[2], m12+m21},
    {m10-m01, m02+m20, m12+m21, qa[3]*qa[3]}};
  int idx = 0; float best = qa[0];
  for (int i = 1; i < 4; ++i) if (qa[i] > best) { best = qa[i]; idx = i; }
  float inv = 1.f / (2.f * fmaxf(qa[idx], 0.1f));
  for (int i = 0; i < 4; ++i) q[i] = cand[idx][i] * inv;
}

__device__ __forceinline__ void quat_mul(const float* a, const float* b, float o[4]) {
  o[0] = a[0]*b[0] - a[1]*b[1] - a[2]*b[2] - a[3]*b[3];
  o[1] = a[0]*b[1] + a[1]*b[0] + a[2]*b[3] - a[3]*b[2];
  o[2] = a[0]*b[2] - a[1]*b[3] + a[2]*b[0] + a[3]*b[1];
  o[3] = a[0]*b[3] + a[1]*b[2] - a[2]*b[1] + a[3]*b[0];
}

__device__ __forceinline__ void quat_to_mat(const float* q, float R[9]) {
  float w=q[0], x=q[1], y=q[2], z=q[3];
  R[0]=1.f-2.f*(y*y+z*z); R[1]=2.f*(x*y-z*w);     R[2]=2.f*(x*z+y*w);
  R[3]=2.f*(x*y+z*w);     R[4]=1.f-2.f*(x*x+z*z); R[5]=2.f*(y*z-x*w);
  R[6]=2.f*(x*z-y*w);     R[7]=2.f*(y*z+x*w);     R[8]=1.f-2.f*(x*x+y*y);
}

__global__ void pose_kernel(PoseArgs pa, float* wsf) {
  __shared__ float sEmb[21];
  __shared__ float sPose[128];
  __shared__ float sIn[384];
  __shared__ float sH[256];
  __shared__ float sHd[128];
  __shared__ float sOut[150];
  __shared__ float sR[2][25][9];
  __shared__ float sT[2][25][3];
  const int t = threadIdx.x;

  if (t == 0) {
    float tv = (float)(*pa.tidx);
    sEmb[0] = tv;
    for (int f = 0; f < 10; ++f) {
      float ang = tv * (float)(1 << f);
      sEmb[1 + 2*f]     = sinf(ang);
      sEmb[1 + 2*f + 1] = cosf(ang);
    }
  }
  __syncthreads();
  if (t < 128) {
    float acc = pa.basis_b[t];
    for (int k = 0; k < 21; ++k) acc += pa.basis_w[t*21 + k] * sEmb[k];
    sPose[t] = acc;
  }
  __syncthreads();

  for (int ci = 0; ci < 2; ++ci) {
    const float* code = (ci == 0) ? (const float*)sPose : pa.rest_code;
    for (int i = 0; i < 8; ++i) {
      int len;
      if (i == 0) {
        if (t < 128) sIn[t] = code[t];
        len = 128;
      } else if (i == 4) {
        if (t < 128) sIn[t] = code[t];
        sIn[128 + t] = sH[t];
        len = 384;
      } else {
        sIn[t] = sH[t];
        len = 256;
      }
      __syncthreads();
      {
        const float* W = pa.w[i];
        float acc = pa.b[i][t];
        for (int k = 0; k < len; ++k) acc += W[t*len + k] * sIn[k];
        float h = fmaxf(acc, 0.f);
        __syncthreads();
        sH[t] = h;
      }
      __syncthreads();
    }
    {
      float acc = pa.bf[t];
      for (int k = 0; k < 256; ++k) acc += pa.wf[t*256 + k] * sH[k];
      sIn[t] = acc;
    }
    __syncthreads();
    if (t < 128) {
      float acc = pa.bd[t];
      for (int k = 0; k < 256; ++k) acc += pa.wd[t*256 + k] * sIn[k];
      sHd[t] = fmaxf(acc, 0.f);
    }
    __syncthreads();
    if (t < 150) {
      float acc = pa.bo[t];
      for (int k = 0; k < 128; ++k) acc += pa.wo[t*128 + k] * sHd[k];
      sOut[t] = acc;
    }
    __syncthreads();
    if (t < 25) {
      float tv[3], wv[3];
      for (int i = 0; i < 3; ++i) { tv[i] = sOut[6*t + i] * 0.1f; wv[i] = sOut[6*t + 3 + i]; }
      float R[9];
      so3_exp(wv, R);
      for (int i = 0; i < 9; ++i) sR[ci][t][i] = R[i];
      for (int i = 0; i < 3; ++i) sT[ci][t][i] = tv[i];
    }
    __syncthreads();
  }

  if (t < 25) {
    const float* Rfw = sR[0][t]; const float* tfw = sT[0][t];
    const float* Rr  = sR[1][t]; const float* tr  = sT[1][t];
    float Rm[9], Tm[3];
    for (int i = 0; i < 3; ++i) {
      for (int j = 0; j < 3; ++j) {
        float a = 0.f;
        for (int k = 0; k < 3; ++k) a += Rr[k*3 + i] * Rfw[k*3 + j];
        Rm[i*3 + j] = a;
      }
      float a = 0.f;
      for (int k = 0; k < 3; ++k) a += Rr[k*3 + i] * (tfw[k] - tr[k]);
      Tm[i] = a;
    }
    const float* bn = pa.bones + t*10;
    float ctr[3];
    for (int i = 0; i < 3; ++i) {
      float a = tr[i];
      for (int j = 0; j < 3; ++j) a += Rr[i*3 + j] * bn[j];
      ctr[i] = a;
    }
    float qR[4], q[4];
    mat_to_quat(Rr, qR);
    quat_mul(qR, bn + 3, q);
    float qn = rsqrtf(q[0]*q[0] + q[1]*q[1] + q[2]*q[2] + q[3]*q[3]);
    for (int i = 0; i < 4; ++i) q[i] *= qn;
    float Rb[9];
    quat_to_mat(q, Rb);
    float* dst = wsf + FBONE + t*27;
    for (int i = 0; i < 3; ++i) dst[i]      = ctr[i];
    for (int i = 0; i < 9; ++i) dst[3 + i]  = Rb[i];
    for (int i = 0; i < 3; ++i) dst[12 + i] = expf(bn[7 + i]);
    for (int i = 0; i < 9; ++i) dst[15 + i] = Rm[i];
    for (int i = 0; i < 3; ++i) dst[24 + i] = Tm[i];
  }
  if (t == 0) wsf[FBONE + 675] = -1000.f * expf(pa.skin_aux[0]);
}

// ======================================================================
// Kernel 3: per-point skin MLP via WMMA + skinning epilogue
//   Register-blocked: MT M-tiles x NT N-tiles of accumulators live in
//   VGPRs; A reused across N-tiles, B reused across M-tiles.
// ======================================================================
template<int KP, int NT, int MT, bool RELU, bool TOF32>
__device__ __forceinline__ void wave_gemm(const _Float16* __restrict__ Wt,
                                          const float* __restrict__ Bias,
                                          const _Float16* A, int aStride,
                                          _Float16* Oh, float* Of, int oStride,
                                          int lane) {
  const int l15  = lane & 15;
  const int koff = (lane & 16) ? 8 : 0;   // K sub-offset per ISA A/B layout
  const int prow = (lane & 16) ? 8 : 0;   // C/D row offset
  v8f acc[MT][NT];
#pragma unroll
  for (int nt = 0; nt < NT; ++nt) {
    float bv = Bias[nt*16 + l15];
#pragma unroll
    for (int m = 0; m < MT; ++m)
#pragma unroll
      for (int r = 0; r < 8; ++r) acc[m][nt][r] = bv;
  }
#pragma unroll
  for (int kt = 0; kt < KP/32; ++kt) {
    const int kb = kt*32 + koff;
    v16h av[MT];
#pragma unroll
    for (int m = 0; m < MT; ++m) {
      const _Float16* arow = A + (m*16 + l15)*aStride + kb;
      v8h a0 = *(const v8h*)(arow);
      v8h a1 = *(const v8h*)(arow + 16);
      av[m] = __builtin_shufflevector(a0, a1, 0,1,2,3,4,5,6,7,8,9,10,11,12,13,14,15);
    }
#pragma unroll
    for (int nt = 0; nt < NT; ++nt) {
      const _Float16* wrow = Wt + (nt*16 + l15)*KP + kb;
      v8h b0 = *(const v8h*)(wrow);
      v8h b1 = *(const v8h*)(wrow + 16);
      v16h bm = __builtin_shufflevector(b0, b1, 0,1,2,3,4,5,6,7,8,9,10,11,12,13,14,15);
#pragma unroll
      for (int m = 0; m < MT; ++m)
        acc[m][nt] = __builtin_amdgcn_wmma_f32_16x16x32_f16(false, av[m], false, bm,
                                                            (short)0, acc[m][nt],
                                                            false, false);
    }
  }
#pragma unroll
  for (int m = 0; m < MT; ++m)
#pragma unroll
    for (int nt = 0; nt < NT; ++nt)
#pragma unroll
      for (int r = 0; r < 8; ++r) {
        float v = acc[m][nt][r];
        if (RELU) v = fmaxf(v, 0.f);
        const int row = m*16 + prow + r;
        if (TOF32) Of[row*oStride + nt*16 + l15] = v;
        else       Oh[row*oStride + nt*16 + l15] = (_Float16)v;
      }
}

#define WAVES 2
#define PTSW  32   // points per wave (MT=2 M-tiles)

__global__ void __launch_bounds__(64)
skin_kernel(const float* __restrict__ xyz, const _Float16* __restrict__ ws16,
            const float* __restrict__ wsf, float* __restrict__ out, int npts) {
  __shared__ __align__(16) _Float16 shIN[WAVES][PTSW*256];  // 32 KB
  __shared__ __align__(16) _Float16 shH0[WAVES][PTSW*64];   //  8 KB
  __shared__ __align__(16) _Float16 shH1[WAVES][PTSW*64];   //  8 KB
  __shared__ __align__(16) float    shSK[WAVES][PTSW*32];   //  8 KB
  __shared__ float shX[WAVES][PTSW][3];
  __shared__ float shBone[FBONE_CNT];

  const int w    = threadIdx.x >> 5;
  const int lane = threadIdx.x & 31;
  const int base = blockIdx.x * (WAVES*PTSW) + w * PTSW;

  for (int i = threadIdx.x; i < FBONE_CNT; i += blockDim.x)
    shBone[i] = wsf[FBONE + i];
  {
    int n = min(base + lane, npts - 1);
    shX[w][lane][0] = xyz[n*3 + 0];
    shX[w][lane][1] = xyz[n*3 + 1];
    shX[w][lane][2] = xyz[n*3 + 2];
  }
  __syncthreads();

  // ---- build embedded input: [x(3) | PE(60) | code(128) | pad->256] ----
  _Float16* IN = shIN[w];
  for (int idx = lane; idx < PTSW*64; idx += 32) {       // zero pad 192..255
    int p = idx >> 6, f = idx & 63;
    IN[p*256 + 192 + f] = (_Float16)0.f;
  }
  IN[lane*256 + 191] = (_Float16)0.f;                    // pad slot 191 (pre-skip)
  for (int idx = lane; idx < PTSW*64; idx += 32) {       // xyz + PE -> 0..62
    int p = idx >> 6, f = idx & 63;
    if (f < 63) {
      float v;
      if (f < 3) v = shX[w][p][f];
      else {
        int k = f - 3;
        int fr = k / 6, rem = k - 6*fr;
        int s = rem / 3, d = rem - 3*s;
        float ang = shX[w][p][d] * (float)(1 << fr);
        v = s ? cosf(ang) : sinf(ang);
      }
      IN[p*256 + f] = (_Float16)v;
    }
  }
  const _Float16* code16 = ws16 + WS_CODE;
  for (int idx = lane; idx < PTSW*128; idx += 32) {      // code -> 63..190
    int p = idx >> 7, k = idx & 127;
    IN[p*256 + 63 + k] = code16[k];
  }
  __syncthreads();

  // ---- MLP: all WMMA ----
  _Float16* H0 = shH0[w];
  _Float16* H1 = shH1[w];
  float*    SK = shSK[w];
  wave_gemm<192, 4, 2, true,  false>(ws16 + WS_W0, wsf + FB0, IN,  256, H0, nullptr, 64, lane);
  wave_gemm< 64, 4, 2, true,  false>(ws16 + WS_W1, wsf + FB1, H0,  64,  H1, nullptr, 64, lane);
  wave_gemm< 64, 4, 2, true,  false>(ws16 + WS_W2, wsf + FB2, H1,  64,  H0, nullptr, 64, lane);
  wave_gemm< 64, 4, 2, true,  false>(ws16 + WS_W3, wsf + FB3, H0,  64,  IN + 191, nullptr, 256, lane); // skip concat
  wave_gemm<256, 4, 2, true,  false>(ws16 + WS_W4, wsf + FB4, IN,  256, H0, nullptr, 64, lane);
  wave_gemm< 64, 4, 2, false, false>(ws16 + WS_WF, wsf + FBF, H0,  64,  H1, nullptr, 64, lane);
  wave_gemm< 64, 2, 2, true,  false>(ws16 + WS_WD, wsf + FBD, H1,  64,  H0, nullptr, 64, lane);
  wave_gemm< 32, 2, 2, false, true >(ws16 + WS_WO, wsf + FBO, H0,  64,  nullptr, SK, 32, lane);

  // ---- skinning epilogue: one lane per point (all 32 lanes active) ----
  {
    const int p = lane;
    const int n = base + p;
    if (n < npts) {
      const float x0 = shX[w][p][0], x1 = shX[w][p][1], x2 = shX[w][p][2];
      const float alpha = shBone[675];
      float lg[25];
      float mx = -1e30f;
#pragma unroll 1
      for (int b = 0; b < 25; ++b) {
        const float* B = shBone + b*27;
        float d0 = B[0] - x0, d1 = B[1] - x1, d2 = B[2] - x2;
        // e = Rb^T d  (einsum 'bji,nbj->nbi')
        float e0 = B[3+0]*d0 + B[3+3]*d1 + B[3+6]*d2;
        float e1 = B[3+1]*d0 + B[3+4]*d1 + B[3+7]*d2;
        float e2 = B[3+2]*d0 + B[3+5]*d1 + B[3+8]*d2;
        float m  = alpha * (B[12]*e0*e0 + B[13]*e1*e1 + B[14]*e2*e2);
        float l  = m + SK[p*32 + b];
        lg[b] = l;
        mx = fmaxf(mx, l);
      }
      float sum = 0.f, o0 = 0.f, o1 = 0.f, o2 = 0.f;
#pragma unroll 1
      for (int b = 0; b < 25; ++b) {
        float e = expf(lg[b] - mx);
        sum += e;
        const float* Rm = shBone + b*27 + 15;
        const float* Tm = shBone + b*27 + 24;
        o0 += e * (Rm[0]*x0 + Rm[1]*x1 + Rm[2]*x2 + Tm[0]);
        o1 += e * (Rm[3]*x0 + Rm[4]*x1 + Rm[5]*x2 + Tm[1]);
        o2 += e * (Rm[6]*x0 + Rm[7]*x1 + Rm[8]*x2 + Tm[2]);
      }
      float inv = 1.f / sum;
      out[n*3 + 0] = o0 * inv;
      out[n*3 + 1] = o1 * inv;
      out[n*3 + 2] = o2 * inv;
    }
  }
}

// ======================================================================
extern "C" void kernel_launch(void* const* d_in, const int* in_sizes, int n_in,
                              void* d_out, int out_size, void* d_ws, size_t ws_size,
                              hipStream_t stream) {
  (void)n_in; (void)out_size; (void)ws_size;
  const float* xyz = (const float*)d_in[0];

  _Float16* ws16 = (_Float16*)d_ws;
  float*    wsf  = (float*)((char*)d_ws + WS_F32_BYTES);

  PrepArgs pr;
  pr.w0 = (const float*)d_in[29]; pr.b0 = (const float*)d_in[30];
  pr.w1 = (const float*)d_in[31]; pr.b1 = (const float*)d_in[32];
  pr.w2 = (const float*)d_in[33]; pr.b2 = (const float*)d_in[34];
  pr.w3 = (const float*)d_in[35]; pr.b3 = (const float*)d_in[36];
  pr.w4 = (const float*)d_in[37]; pr.b4 = (const float*)d_in[38];
  pr.wf = (const float*)d_in[39]; pr.bf = (const float*)d_in[40];
  pr.wd = (const float*)d_in[41]; pr.bd = (const float*)d_in[42];
  pr.wo = (const float*)d_in[43]; pr.bo = (const float*)d_in[44];
  pr.rest = (const float*)d_in[4];

  PoseArgs pa;
  pa.tidx      = (const int*)d_in[1];
  pa.bones     = (const float*)d_in[2];
  pa.skin_aux  = (const float*)d_in[3];
  pa.rest_code = (const float*)d_in[4];
  pa.basis_w   = (const float*)d_in[5];
  pa.basis_b   = (const float*)d_in[6];
  for (int i = 0; i < 8; ++i) {
    pa.w[i] = (const float*)d_in[7 + 2*i];
    pa.b[i] = (const float*)d_in[8 + 2*i];
  }
  pa.wf = (const float*)d_in[23]; pa.bf = (const float*)d_in[24];
  pa.wd = (const float*)d_in[25]; pa.bd = (const float*)d_in[26];
  pa.wo = (const float*)d_in[27]; pa.bo = (const float*)d_in[28];

  const int npts = in_sizes[0] / 3;

  prep_kernel<<<32, 256, 0, stream>>>(pr, ws16, wsf);
  pose_kernel<<<1, 256, 0, stream>>>(pa, wsf);
  const int blocks = (npts + WAVES*PTSW - 1) / (WAVES*PTSW);
  skin_kernel<<<blocks, WAVES*32, 0, stream>>>(xyz, ws16, wsf, (float*)d_out, npts);
}
